// GATLayer_2516850835555
// MI455X (gfx1250) — matrix operand order repaired and checked by reading, not compiled, not run on hardware
//
#include <hip/hip_runtime.h>
#include <hip/hip_bf16.h>
#include <math.h>

typedef __attribute__((ext_vector_type(16))) _Float16 v16h;
typedef __attribute__((ext_vector_type(8)))  float    v8f;

constexpr int B_ = 2, N_ = 4096, IN_ = 64, H_ = 4, D_ = 32;
constexpr float NEG_SLOPE = 0.1f;
constexpr float NEG_INF_C = -9e15f;
constexpr float LOG2E = 1.4426950408889634f;
constexpr int NT16 = N_ / 16;   // 256 row/key tiles of 16
constexpr int NT32 = N_ / 32;   // 128 key tiles of 32

// Single-instruction float max (avoids llvm.canonicalize max(x,x) spam around fmaxf)
__device__ __forceinline__ float vmax(float a, float b) {
    float r;
    asm("v_max_num_f32 %0, %1, %2" : "=v"(r) : "v"(a), "v"(b));
    return r;
}

// ---------- 16-lane butterfly reductions (C-layout: 16 lanes share a row) ----------
__device__ __forceinline__ float rmax16(float v) {
    #pragma unroll
    for (int m = 1; m < 16; m <<= 1) v = vmax(v, __shfl_xor(v, m, 32));
    return v;
}
__device__ __forceinline__ float rsum16(float v) {
    #pragma unroll
    for (int m = 1; m < 16; m <<= 1) v += __shfl_xor(v, m, 32);
    return v;
}

// ---------------- Kernel 0: additive mask  adjneg = (adj==0)? -9e15 : 0 ----------------
__global__ void k_mask(const float* __restrict__ adj, float* __restrict__ adjneg) {
    const long i = (long)blockIdx.x * blockDim.x + threadIdx.x;
    adjneg[i] = (adj[i] == 0.f) ? NEG_INF_C : 0.f;
}

// ---------------- Kernel 1: xp = x @ fc_w^T + fc_b  ([B*N, 128]) ----------------
__global__ void k_proj(const float* __restrict__ x, const float* __restrict__ fc_w,
                       const float* __restrict__ fc_b, float* __restrict__ xp) {
    __shared__ float xr[IN_];
    const int bn = blockIdx.x;        // b*N + n
    const int j  = threadIdx.x;       // 0..127 output feature
    if (j < IN_) xr[j] = x[(long)bn * IN_ + j];
    __syncthreads();
    float acc = fc_b[j];
    #pragma unroll
    for (int k = 0; k < IN_; ++k) acc += xr[k] * fc_w[j * IN_ + k];
    xp[(long)bn * (H_ * D_) + j] = acc;
}

// ---------------- Kernel 2: per-head transforms + WMMA-layout f16 packing ----------------
__global__ void k_pack(const float* __restrict__ xp,
                       const float* __restrict__ Q1, const float* __restrict__ Q2,
                       const float* __restrict__ Km, const float* __restrict__ Vm,
                       _Float16* __restrict__ qbuf, _Float16* __restrict__ kbuf,
                       _Float16* __restrict__ vbuf) {
    __shared__ float sQ[D_ * D_], sK[D_ * D_], sV[D_ * D_];
    const long idx = (long)blockIdx.x * blockDim.x + threadIdx.x;
    const int n = (int)(idx % N_);
    const int h = (int)((idx / N_) % H_);
    const int b = (int)(idx / ((long)N_ * H_));
    for (int i = threadIdx.x; i < D_ * D_; i += blockDim.x) {
        sQ[i] = Q1[h * D_ * D_ + i] + Q2[h * D_ * D_ + i];
        sK[i] = Km[h * D_ * D_ + i];
        sV[i] = Vm[h * D_ * D_ + i];
    }
    __syncthreads();

    float xv[D_];
    #pragma unroll
    for (int d = 0; d < D_; ++d)
        xv[d] = xp[((long)b * N_ + n) * (H_ * D_) + h * D_ + d];

    const int bh = b * H_ + h;

    // Qt (post-leaky): A-layout.  lane = m + 16*((k>>3)&1), elem = ((k>>4)<<3)|(k&7)
    {
        const long base = ((long)bh * NT16 + (n >> 4)) * 512;
        const int m = n & 15;
        #pragma unroll 1
        for (int e = 0; e < D_; ++e) {
            float a = 0.f;
            #pragma unroll
            for (int d = 0; d < D_; ++d) a += xv[d] * sQ[d * D_ + e];
            a = vmax(a, NEG_SLOPE * a);                     // leaky
            qbuf[base + (m + 16 * ((e >> 3) & 1)) * 16 + (((e >> 4) << 3) | (e & 7))] = (_Float16)a;
        }
    }
    // Kt: B-layout (K = feature e, col = node).  lane = c + 16*((e>>3)&1), elem = ((e>>4)<<3)|(e&7)
    {
        const long base = ((long)bh * NT16 + (n >> 4)) * 512;
        const int c = n & 15;
        #pragma unroll 1
        for (int e = 0; e < D_; ++e) {
            float a = 0.f;
            #pragma unroll
            for (int d = 0; d < D_; ++d) a += xv[d] * sK[d * D_ + e];
            kbuf[base + (c + 16 * ((e >> 3) & 1)) * 16 + (((e >> 4) << 3) | (e & 7))] = (_Float16)a;
        }
    }
    // Vt: B-layout (K = key row r, col = feature d), two 16-col halves per key tile.
    {
        const int r = n & 31;
        const long t = n >> 5;
        #pragma unroll 1
        for (int d = 0; d < D_; ++d) {
            float a = 0.f;
            #pragma unroll
            for (int dd = 0; dd < D_; ++dd) a += xv[dd] * sV[dd * D_ + d];
            const long base = (((long)bh * NT32 + t) * 2 + (d >> 4)) * 512;
            vbuf[base + ((d & 15) + 16 * ((r >> 3) & 1)) * 16 + (((r >> 4) << 3) | (r & 7))] = (_Float16)a;
        }
    }
}

// ---------------- Kernel 3: fused flash-attention over adj-masked leaky scores ----------------
__global__ void __launch_bounds__(256)
k_attn(const _Float16* __restrict__ qbuf, const _Float16* __restrict__ kbuf,
       const _Float16* __restrict__ vbuf, const float* __restrict__ adjneg,
       float* __restrict__ out) {
    __shared__ _Float16 lp[8 * 512];          // 1KB per wave: P-tile C->A re-swizzle buffer
    const int wave = threadIdx.x >> 5;
    const int lane = threadIdx.x & 31;
    const int hi   = lane >> 4;               // C-layout: high half-wave holds rows m+8
    const int cl   = lane & 15;               // C-layout: column within tile
    const int gw   = blockIdx.x * 8 + wave;
    const int rt   = gw % NT16;
    const int h    = (gw / NT16) % H_;
    const int b    = gw / (NT16 * H_);
    const int bh   = b * H_ + h;

    // Query tile (A-layout), loaded once, reused across all 128 key iterations
    const v16h aq = *reinterpret_cast<const v16h*>(qbuf + ((long)bh * NT16 + rt) * 512 + lane * 16);

    v8f acc0 = {}, acc1 = {};
    float mrow[8], lpart[8];
    #pragma unroll
    for (int g = 0; g < 8; ++g) { mrow[g] = -INFINITY; lpart[g] = 0.f; }

    _Float16* myp = lp + wave * 512;
    const float* adjr = adjneg + (long)(rt * 16 + 8 * hi) * N_ + cl;  // +g*N_ per row

    for (int kb = 0; kb < NT32; ++kb) {
        // --- scores for 32 keys: two 16-key WMMAs, K-dim = D = 32 ---
        v8f s0, s1;
        {
            const v16h bk = *reinterpret_cast<const v16h*>(
                kbuf + ((long)bh * NT16 + 2 * kb) * 512 + lane * 16);
            v8f z = {};
            s0 = __builtin_amdgcn_wmma_f32_16x16x32_f16(false, aq, false, bk, (short)0, z, false, false);
        }
        {
            const v16h bk = *reinterpret_cast<const v16h*>(
                kbuf + ((long)bh * NT16 + 2 * kb + 1) * 512 + lane * 16);
            v8f z = {};
            s1 = __builtin_amdgcn_wmma_f32_16x16x32_f16(false, aq, false, bk, (short)0, z, false, false);
        }

        __builtin_prefetch(adjr + (kb + 4) * 32, 0, 1);   // gfx1250 global_prefetch_b8

        // --- leaky + additive mask + per-row online max ---
        float vv0[8], vv1[8];
        bool same = true;
        #pragma unroll
        for (int g = 0; g < 8; ++g) {
            const float a0 = adjr[(long)g * N_ + kb * 32];
            const float a1 = adjr[(long)g * N_ + kb * 32 + 16];
            float t0 = s0[g]; t0 = vmax(t0, NEG_SLOPE * t0) + a0;   // leaky then mask-add
            float t1 = s1[g]; t1 = vmax(t1, NEG_SLOPE * t1) + a1;
            vv0[g] = t0; vv1[g] = t1;
            const float mn = vmax(mrow[g], rmax16(vmax(t0, t1)));
            same = same && (mn == mrow[g]);
            vv0[g] = t0;                 // keep arrays live
            mrow[g] = same ? mrow[g] : mrow[g];  // no-op; real update below
            // stash new max in t-slot: reuse vv1 path is unchanged
            // (store candidate max separately)
            s0[g] = mn;                  // recycle s0 as mne[] storage
        }

        // --- rare path: a row max moved -> rescale running state in place ---
        if (!__all((int)same)) {
            #pragma unroll
            for (int g = 0; g < 8; ++g) {
                const float mn = s0[g];
                const float sc = __builtin_amdgcn_exp2f((mrow[g] - mn) * LOG2E);
                mrow[g] = mn;
                lpart[g] *= sc;
                acc0[g] *= sc;
                acc1[g] *= sc;
            }
        }

        // --- common path: probabilities, deferred row-sum, P -> LDS (A-layout) ---
        #pragma unroll
        for (int g = 0; g < 8; ++g) {
            const float c  = -mrow[g] * LOG2E;
            const float p0 = __builtin_amdgcn_exp2f(fmaf(vv0[g], LOG2E, c));
            const float p1 = __builtin_amdgcn_exp2f(fmaf(vv1[g], LOG2E, c));
            lpart[g] += p0 + p1;
            const int m = g + 8 * hi;
            myp[(m + 16 * ((cl >> 3) & 1)) * 16 + (cl & 7)]              = (_Float16)p0;
            myp[(m + 16 * (((16 + cl) >> 3) & 1)) * 16 + (8 | (cl & 7))] = (_Float16)p1;
        }

        __asm volatile("" ::: "memory");   // DS ops are in-order within a wave; block IR reorder
        const v16h ap = *reinterpret_cast<const v16h*>(myp + lane * 16);
        __asm volatile("" ::: "memory");

        // --- O += P @ Vt (two 16-col halves of D=32) ---
        {
            const v16h bv0 = *reinterpret_cast<const v16h*>(
                vbuf + (((long)bh * NT32 + kb) * 2 + 0) * 512 + lane * 16);
            acc0 = __builtin_amdgcn_wmma_f32_16x16x32_f16(false, ap, false, bv0, (short)0, acc0, false, false);
            const v16h bv1 = *reinterpret_cast<const v16h*>(
                vbuf + (((long)bh * NT32 + kb) * 2 + 1) * 512 + lane * 16);
            acc1 = __builtin_amdgcn_wmma_f32_16x16x32_f16(false, ap, false, bv1, (short)0, acc1, false, false);
        }
    }

    // --- finalize: one deferred row-sum reduction, O/l, ReLU, store [B,N,H,D] ---
    #pragma unroll
    for (int g = 0; g < 8; ++g) {
        const int row = rt * 16 + g + 8 * hi;
        const float inv = 1.f / rsum16(lpart[g]);
        float* orow = out + (((long)b * N_ + row) * H_ + h) * D_;
        orow[cl]      = vmax(acc0[g] * inv, 0.f);
        orow[16 + cl] = vmax(acc1[g] * inv, 0.f);
    }
}

// ---------------- host-side launcher ----------------
extern "C" void kernel_launch(void* const* d_in, const int* in_sizes, int n_in,
                              void* d_out, int out_size, void* d_ws, size_t ws_size,
                              hipStream_t stream) {
    (void)in_sizes; (void)n_in; (void)out_size; (void)ws_size;
    // setup_inputs order: vt, x, adj, fc_w, fc_b, Q1, Q2, K, V  (vt unused by reference)
    const float* x    = (const float*)d_in[1];
    const float* adj  = (const float*)d_in[2];
    const float* fc_w = (const float*)d_in[3];
    const float* fc_b = (const float*)d_in[4];
    const float* Q1   = (const float*)d_in[5];
    const float* Q2   = (const float*)d_in[6];
    const float* Km   = (const float*)d_in[7];
    const float* Vm   = (const float*)d_in[8];
    float* out = (float*)d_out;

    char* ws = (char*)d_ws;
    float*    xp     = (float*)(ws);                                 // 4 MB
    _Float16* qbuf   = (_Float16*)(ws + (size_t)4  * 1024 * 1024);   // 2 MB
    _Float16* kbuf   = (_Float16*)(ws + (size_t)6  * 1024 * 1024);   // 2 MB
    _Float16* vbuf   = (_Float16*)(ws + (size_t)8  * 1024 * 1024);   // 2 MB
    float*    adjneg = (float*)(ws + (size_t)10 * 1024 * 1024);      // 64 MB

    k_mask<<<(N_ * (long)N_) / 256, 256, 0, stream>>>(adj, adjneg);
    k_proj<<<B_ * N_, 128, 0, stream>>>(x, fc_w, fc_b, xp);
    k_pack<<<(B_ * H_ * N_) / 256, 256, 0, stream>>>(xp, Q1, Q2, Km, Vm, qbuf, kbuf, vbuf);
    k_attn<<<(B_ * H_ * NT16) / 8, 256, 0, stream>>>(qbuf, kbuf, vbuf, adjneg, out);
}
